// InfiniteHoloGPT_88536455839982
// MI455X (gfx1250) — compile-verified, matrix operand order checked by reading, not run on hardware
//
#include <hip/hip_runtime.h>
#include <hip/hip_bf16.h>
#include <math.h>

// ---------------- model constants (from reference) ----------------
#define Vv   32000
#define Ee   512
#define Mm   512
#define Ll   4
#define Hh   16
#define HDd  32
#define Bb   2
#define Tt   1024
#define BT   (Bb*Tt)      // 2048 token rows
#define FF   (4*Ee)       // 2048 ffn hidden

typedef __bf16 bf16;
typedef __attribute__((ext_vector_type(16))) __bf16 v16bf;
typedef __attribute__((ext_vector_type(8)))  __bf16 v8bf;
typedef __attribute__((ext_vector_type(8)))  float  v8f;
typedef __attribute__((ext_vector_type(4)))  float  vf4;
typedef __attribute__((ext_vector_type(4)))  int    v4i;

// ---------------- CDNA5 async global->LDS copy (guarded) ----------------
#if __has_builtin(__builtin_amdgcn_global_load_async_to_lds_b128)
#define HAS_ASYNC_LDS 1
typedef __attribute__((address_space(1))) v4i* gv4i_p;   // global 16B vector ptr
typedef __attribute__((address_space(3))) v4i* lv4i_p;   // LDS    16B vector ptr
#else
#define HAS_ASYNC_LDS 0
#endif

__device__ __forceinline__ void wait_asynccnt0() {
#if __has_builtin(__builtin_amdgcn_s_wait_asynccnt)
  __builtin_amdgcn_s_wait_asynccnt(0);
#else
  asm volatile("s_wait_asynccnt 0" ::: "memory");
#endif
}

// ---------------- f32 -> bf16 convert (weights, once per launch) ----------------
__global__ __launch_bounds__(256) void cvt_f32_bf16(const float* __restrict__ src,
                                                    bf16* __restrict__ dst, size_t n) {
  size_t i = (size_t)blockIdx.x * blockDim.x + threadIdx.x;
  size_t stride = (size_t)gridDim.x * blockDim.x;
  for (; i < n; i += stride) dst[i] = (bf16)src[i];
}

// ---------------- embedding gather: x = center[idx]  (offset cancels exactly) ----
__global__ __launch_bounds__(128) void embed_kernel(const int* __restrict__ idx,
                                                    const float* __restrict__ center,
                                                    float* __restrict__ x) {
  int bt = blockIdx.x;
  int tok = idx[bt];
  ((vf4*)(x + (size_t)bt * Ee))[threadIdx.x] =
      ((const vf4*)(center + (size_t)tok * Ee))[threadIdx.x];
}

// ---------------- LayerNorm over E=512, writes bf16 (GEMM A operand) ------------
__global__ __launch_bounds__(128) void layernorm_kernel(const float* __restrict__ x,
                                                        const float* __restrict__ g,
                                                        const float* __restrict__ b,
                                                        bf16* __restrict__ out) {
  __shared__ float red[8];
  int row = blockIdx.x, tid = threadIdx.x;
  const float* xr = x + (size_t)row * Ee;
  vf4 v = ((const vf4*)xr)[tid];
  float e0 = v.x, e1 = v.y, e2 = v.z, e3 = v.w;
  float s  = e0 + e1 + e2 + e3;
  float ss = e0*e0 + e1*e1 + e2*e2 + e3*e3;
  #pragma unroll
  for (int o = 16; o > 0; o >>= 1) { s += __shfl_xor(s, o, 32); ss += __shfl_xor(ss, o, 32); }
  int wave = tid >> 5, lane = tid & 31;
  if (lane == 0) { red[wave] = s; red[4 + wave] = ss; }
  __syncthreads();
  s  = red[0] + red[1] + red[2] + red[3];
  ss = red[4] + red[5] + red[6] + red[7];
  float mean = s * (1.0f / Ee);
  float var  = ss * (1.0f / Ee) - mean * mean;
  float rstd = rsqrtf(var + 1e-5f);
  float vv[4] = {e0, e1, e2, e3};
  #pragma unroll
  for (int c = 0; c < 4; ++c) {
    int e = tid * 4 + c;
    out[(size_t)row * Ee + e] = (bf16)((vv[c] - mean) * rstd * g[e] + b[e]);
  }
}

// ---------------- WMMA GEMM: out[M,N] = epi(alpha*A[M,K]@W[N,K]^T + bias) -------
// A row-major bf16 (K contiguous), W row-major bf16 [N,K] (einsum 'btc,oc->bto').
// Block tile 64x64, K-step 32, 8 waves: 4 row-groups x 2 col-groups, 2 WMMA/wave/step.
#define TM 64
#define TN 64
#define TK 32
#define LDST 40   // padded LDS k-stride (80B rows -> conflict-free b128 reads)

// Fragment per ISA 7.12.2 (16-bit A 16x32): lane<16 holds K {kh*8..+7, 16+kh*8..+7}
__device__ __forceinline__ v16bf load_frag(const bf16* base, int row, int kh) {
  union { v16bf v; v8bf h[2]; } u;
  const bf16* p = base + row * LDST + kh * 8;
  u.h[0] = *(const v8bf*)p;          // elements 0..7  -> K = kh*8 .. +7
  u.h[1] = *(const v8bf*)(p + 16);   // elements 8..15 -> K = 16 + kh*8 .. +7
  return u.v;
}

__device__ __forceinline__ float gelu_exact(float x) {
  return 0.5f * x * (1.0f + erff(x * 0.70710678118654752f));
}

__global__ __launch_bounds__(256)
void gemm_wmma_kernel(const bf16* __restrict__ A, const bf16* __restrict__ W,
                      const float* __restrict__ bias,
                      const float* resid, const float* __restrict__ alpha_ptr,
                      float alpha_const, float* outF, bf16* outH,
                      int Mdim, int Ndim, int Kdim, int do_gelu, int nt_store) {
  __shared__ __align__(16) bf16 As[TM * LDST];
  __shared__ __align__(16) bf16 Ws[TN * LDST];
  const int tid  = threadIdx.x;
  const int wave = tid >> 5, lane = tid & 31;
  const int rg = wave & 3;        // row group: rows rg*16..+15
  const int cg = wave >> 2;       // col group: cols cg*32..+31
  const int m0 = blockIdx.y * TM;
  const int n0 = blockIdx.x * TN;

  v8f acc0 = {}; v8f acc1 = {};

  const int lrow = tid >> 2;            // 0..63
  const int lcol = (tid & 3) * 8;       // 0,8,16,24 (16B chunks)

  for (int k0 = 0; k0 < Kdim; k0 += TK) {
    __syncthreads();
    const bf16* ga = A + (size_t)(m0 + lrow) * Kdim + k0 + lcol;
    const bf16* gw = W + (size_t)(n0 + lrow) * Kdim + k0 + lcol;
#if HAS_ASYNC_LDS
    // CDNA5 async DMA: global -> LDS directly (ASYNCcnt), no VGPR staging.
    __builtin_amdgcn_global_load_async_to_lds_b128(
        (gv4i_p)(void*)(bf16*)ga, (lv4i_p)&As[lrow * LDST + lcol], 0, 0);
    __builtin_amdgcn_global_load_async_to_lds_b128(
        (gv4i_p)(void*)(bf16*)gw, (lv4i_p)&Ws[lrow * LDST + lcol], 0, 0);
#else
    *(vf4*)&As[lrow * LDST + lcol] = *(const vf4*)ga;
    *(vf4*)&Ws[lrow * LDST + lcol] = *(const vf4*)gw;
#endif
    if (k0 + TK < Kdim) {               // CDNA5 global_prefetch_b8 of next K tile
      __builtin_prefetch(ga + TK, 0, 1);
      __builtin_prefetch(gw + TK, 0, 1);
    }
#if HAS_ASYNC_LDS
    wait_asynccnt0();                   // this wave's LDS writes landed
#endif
    __syncthreads();                    // all waves' tiles visible

    const int r  = rg * 16 + (lane & 15);
    const int kh = lane >> 4;
    v16bf afrag = load_frag(As, r, kh);
    v16bf b0    = load_frag(Ws, cg * 32 + (lane & 15), kh);
    v16bf b1    = load_frag(Ws, cg * 32 + 16 + (lane & 15), kh);
    acc0 = __builtin_amdgcn_wmma_f32_16x16x32_bf16(false, afrag, false, b0,
                                                   (short)0, acc0, false, false);
    acc1 = __builtin_amdgcn_wmma_f32_16x16x32_bf16(false, afrag, false, b1,
                                                   (short)0, acc1, false, false);
  }

  const float alpha = alpha_const * (alpha_ptr ? alpha_ptr[0] : 1.0f);
  // C/D layout: VGPR r -> (M = r + 8*(lane>=16), N = lane&15)
  const int rowb = m0 + rg * 16 + (lane >> 4) * 8;
  const int colb = n0 + cg * 32 + (lane & 15);
  #pragma unroll
  for (int r = 0; r < 8; ++r) {
    const int row = rowb + r;
    float v0 = acc0[r] * alpha;
    float v1 = acc1[r] * alpha;
    if (bias) { v0 += bias[colb]; v1 += bias[colb + 16]; }
    if (do_gelu) { v0 = gelu_exact(v0); v1 = gelu_exact(v1); }
    size_t o0 = (size_t)row * Ndim + colb;
    size_t o1 = o0 + 16;
    if (resid) { v0 += resid[o0]; v1 += resid[o1]; }
    if (outF) {
      if (nt_store) {  // logits: 262MB stream, written once -> bypass caches (TH=NT)
        __builtin_nontemporal_store(v0, &outF[o0]);
        __builtin_nontemporal_store(v1, &outF[o1]);
      } else {
        outF[o0] = v0; outF[o1] = v1;
      }
    } else {
      outH[o0] = (bf16)v0; outH[o1] = (bf16)v1;
    }
  }
}

// ---------------- gate projections: gw/gf [B,T,H] --------------------------------
__global__ __launch_bounds__(256)
void gates_kernel(const bf16* __restrict__ xn, const float* __restrict__ wgw,
                  const float* __restrict__ bgw, const float* __restrict__ wgf,
                  const float* __restrict__ bgf, const float* __restrict__ rarity,
                  float* __restrict__ gw, float* __restrict__ gf) {
  const int bt = blockIdx.x;
  const int wave = threadIdx.x >> 5, lane = threadIdx.x & 31;
  const bf16* xrow = xn + (size_t)bt * Ee;
  const float rar = rarity[bt];
  #pragma unroll
  for (int r = 0; r < 4; ++r) {
    const int d = wave * 4 + r;          // 0..31 : 16 gw dots then 16 gf dots
    const int h = d & 15;
    const float* wrow = (d < 16 ? wgw : wgf) + (size_t)h * Ee;
    float s = 0.0f;
    for (int e = lane; e < Ee; e += 32) s = fmaf((float)xrow[e], wrow[e], s);
    #pragma unroll
    for (int o = 16; o > 0; o >>= 1) s += __shfl_xor(s, o, 32);
    if (lane == 0) {
      if (d < 16) {
        float sig = 1.0f / (1.0f + expf(-(s + bgw[h])));
        gw[(size_t)bt * Hh + h] = sig * (0.5f + 0.5f * rar);
      } else {
        float sig = 1.0f / (1.0f + expf(-(s + bgf[h])));
        gf[(size_t)bt * Hh + h] = sig * (0.9f + 0.1f * (1.0f - rar));
      }
    }
  }
}

// ---------------- holographic scan: one wave per (b,h), mem 32x32 in regs --------
// lane i owns memory row i: mem[i][j] for j=0..31. Per step:
//   readout_i = sum_j mem[i][j]*q[j]   (read BEFORE write)
//   mem[i][j] = gf*mem[i][j] + gw*v[i]*k[j]
__global__ __launch_bounds__(32)
void holo_scan_kernel(const float* __restrict__ kb, const float* __restrict__ qb,
                      const float* __restrict__ vb, const float* __restrict__ gw,
                      const float* __restrict__ gf, bf16* __restrict__ attn_out,
                      float* __restrict__ states_out /* [B,H,32,32] */) {
  const int bh = blockIdx.x;                 // b*H + h
  const int b = bh / Hh, h = bh % Hh;
  const int lane = threadIdx.x;              // row i
  float mem[32];
  #pragma unroll
  for (int j = 0; j < 32; ++j) mem[j] = 0.0f;

  for (int t = 0; t < Tt; ++t) {
    const size_t base = ((size_t)(b * Tt + t) * Mm) + (size_t)h * HDd + lane;
    float kv = kb[base], qv = qb[base], vv = vb[base];
    float ks = kv * kv, qs = qv * qv;
    #pragma unroll
    for (int o = 16; o > 0; o >>= 1) { ks += __shfl_xor(ks, o, 32); qs += __shfl_xor(qs, o, 32); }
    kv *= 1.0f / fmaxf(sqrtf(ks), 1e-12f);   // _l2norm
    qv *= 1.0f / fmaxf(sqrtf(qs), 1e-12f);
    vv = tanhf(vv);
    const size_t gi = (size_t)(b * Tt + t) * Hh + h;
    const float gfv = gf[gi];
    const float wvi = gw[gi] * vv;           // gw * v[i]  (lane-local)
    float ro = 0.0f;
    #pragma unroll
    for (int j = 0; j < 32; ++j) {
      const float qj = __shfl(qv, j, 32);
      const float kj = __shfl(kv, j, 32);
      ro = fmaf(mem[j], qj, ro);             // read old memory
      mem[j] = fmaf(gfv, mem[j], wvi * kj);  // then decay + write
    }
    attn_out[base] = (bf16)ro;
  }
  #pragma unroll
  for (int j = 0; j < 32; ++j)
    states_out[((size_t)bh * 32 + lane) * 32 + j] = mem[j];
}

// ---------------- host-side launch helpers ---------------------------------------
static void launch_gemm(hipStream_t s, const bf16* A, const bf16* W, const float* bias,
                        const float* resid, const float* alpha_ptr, float alpha_const,
                        float* outF, bf16* outH, int M, int N, int K, int do_gelu,
                        int nt_store = 0) {
  dim3 grid(N / TN, M / TM), block(256);
  gemm_wmma_kernel<<<grid, block, 0, s>>>(A, W, bias, resid, alpha_ptr, alpha_const,
                                          outF, outH, M, N, K, do_gelu, nt_store);
}

static void launch_cvt(hipStream_t s, const float* src, bf16* dst, size_t n) {
  int blocks = (int)((n + 255) / 256); if (blocks > 2048) blocks = 2048;
  cvt_f32_bf16<<<blocks, 256, 0, s>>>(src, dst, n);
}

extern "C" void kernel_launch(void* const* d_in, const int* in_sizes, int n_in,
                              void* d_out, int out_size, void* d_ws, size_t ws_size,
                              hipStream_t stream) {
  (void)in_sizes; (void)n_in; (void)out_size; (void)ws_size;
  // setup_inputs order: idx, rarity, then params dict in insertion order.
  const int*   idx       = (const int*)  d_in[0];
  const float* rarity    = (const float*)d_in[1];
  const float* center    = (const float*)d_in[2];
  /* d_in[3] 'offset' unused: x = ((c-o)+(c+o))/2 == c exactly */
  const float* lscale    = (const float*)d_in[4];
  const float* wk = (const float*)d_in[5];  const float* bk = (const float*)d_in[6];
  const float* wq = (const float*)d_in[7];  const float* bq = (const float*)d_in[8];
  const float* wv = (const float*)d_in[9];  const float* bv = (const float*)d_in[10];
  const float* wo = (const float*)d_in[11]; const float* bo = (const float*)d_in[12];
  const float* wgw = (const float*)d_in[13]; const float* bgw = (const float*)d_in[14];
  const float* wgf = (const float*)d_in[15]; const float* bgf = (const float*)d_in[16];
  const float* ln1g = (const float*)d_in[17]; const float* ln1b = (const float*)d_in[18];
  const float* ln2g = (const float*)d_in[19]; const float* ln2b = (const float*)d_in[20];
  const float* fw1 = (const float*)d_in[21]; const float* fb1 = (const float*)d_in[22];
  const float* fw2 = (const float*)d_in[23]; const float* fb2 = (const float*)d_in[24];
  const float* lnfg = (const float*)d_in[25]; const float* lnfb = (const float*)d_in[26];

  float* out    = (float*)d_out;
  float* logits = out;                                   // [B,T,V]
  float* states = out + (size_t)BT * Vv;                 // [L,B,H,32,32]

  // -------- workspace bump allocator --------
  char* ws = (char*)d_ws;
  auto alloc = [&](size_t bytes) -> void* {
    void* p = (void*)ws; ws += (bytes + 255) & ~(size_t)255; return p;
  };
  float* x      = (float*)alloc((size_t)BT * Ee * 4);        // residual stream
  bf16*  xn     = (bf16*) alloc((size_t)BT * Ee * 2);        // LN output (GEMM A)
  float* kbuf   = (float*)alloc((size_t)BT * Mm * 4);
  float* qbuf   = (float*)alloc((size_t)BT * Mm * 4);
  float* vbuf   = (float*)alloc((size_t)BT * Mm * 4);
  float* gwb    = (float*)alloc((size_t)BT * Hh * 4);
  float* gfb    = (float*)alloc((size_t)BT * Hh * 4);
  bf16*  attn   = (bf16*) alloc((size_t)BT * Mm * 2);
  bf16*  hbuf   = (bf16*) alloc((size_t)BT * FF * 2);
  bf16*  wk_h   = (bf16*) alloc((size_t)Ll * Mm * Ee * 2);
  bf16*  wq_h   = (bf16*) alloc((size_t)Ll * Mm * Ee * 2);
  bf16*  wv_h   = (bf16*) alloc((size_t)Ll * Mm * Ee * 2);
  bf16*  wo_h   = (bf16*) alloc((size_t)Ll * Ee * Mm * 2);
  bf16*  w1_h   = (bf16*) alloc((size_t)Ll * FF * Ee * 2);
  bf16*  w2_h   = (bf16*) alloc((size_t)Ll * Ee * FF * 2);
  bf16*  ctr_h  = (bf16*) alloc((size_t)Vv * Ee * 2);

  // -------- weight conversion (bf16 operands for WMMA) --------
  launch_cvt(stream, wk, wk_h, (size_t)Ll * Mm * Ee);
  launch_cvt(stream, wq, wq_h, (size_t)Ll * Mm * Ee);
  launch_cvt(stream, wv, wv_h, (size_t)Ll * Mm * Ee);
  launch_cvt(stream, wo, wo_h, (size_t)Ll * Ee * Mm);
  launch_cvt(stream, fw1, w1_h, (size_t)Ll * FF * Ee);
  launch_cvt(stream, fw2, w2_h, (size_t)Ll * Ee * FF);
  launch_cvt(stream, center, ctr_h, (size_t)Vv * Ee);

  // -------- embedding --------
  embed_kernel<<<BT, 128, 0, stream>>>(idx, center, x);

  // -------- transformer blocks --------
  for (int i = 0; i < Ll; ++i) {
    const size_t wME = (size_t)i * Mm * Ee;
    const size_t wFE = (size_t)i * FF * Ee;
    // ln1
    layernorm_kernel<<<BT, 128, 0, stream>>>(x, ln1g + (size_t)i * Ee, ln1b + (size_t)i * Ee, xn);
    // k/q/v projections (f32 out for scan preprocessing)
    launch_gemm(stream, xn, wk_h + wME, bk + (size_t)i * Mm, nullptr, nullptr, 1.0f,
                kbuf, nullptr, BT, Mm, Ee, 0);
    launch_gemm(stream, xn, wq_h + wME, bq + (size_t)i * Mm, nullptr, nullptr, 1.0f,
                qbuf, nullptr, BT, Mm, Ee, 0);
    launch_gemm(stream, xn, wv_h + wME, bv + (size_t)i * Mm, nullptr, nullptr, 1.0f,
                vbuf, nullptr, BT, Mm, Ee, 0);
    // gates
    gates_kernel<<<BT, 256, 0, stream>>>(xn, wgw + (size_t)i * Hh * Ee, bgw + (size_t)i * Hh,
                                         wgf + (size_t)i * Hh * Ee, bgf + (size_t)i * Hh,
                                         rarity, gwb, gfb);
    // sequential scan over T (32 independent (b,h) chains)
    holo_scan_kernel<<<Bb * Hh, 32, 0, stream>>>(kbuf, qbuf, vbuf, gwb, gfb, attn,
                                                 states + (size_t)i * Bb * Hh * HDd * HDd);
    // output proj + residual (in-place on x)
    launch_gemm(stream, attn, wo_h + wME, bo + (size_t)i * Ee, x, nullptr, 1.0f,
                x, nullptr, BT, Ee, Mm, 0);
    // ffn
    layernorm_kernel<<<BT, 128, 0, stream>>>(x, ln2g + (size_t)i * Ee, ln2b + (size_t)i * Ee, xn);
    launch_gemm(stream, xn, w1_h + wFE, fb1 + (size_t)i * FF, nullptr, nullptr, 1.0f,
                nullptr, hbuf, BT, FF, Ee, /*gelu*/1);
    launch_gemm(stream, hbuf, w2_h + wFE, fb2 + (size_t)i * Ee, x, nullptr, 1.0f,
                x, nullptr, BT, Ee, FF, 0);
  }

  // -------- final LN + logits --------
  layernorm_kernel<<<BT, 128, 0, stream>>>(x, lnfg, lnfb, xn);
  const float inv_sqrt_e = 0.044194173824159216f;  // 1/sqrt(512)
  launch_gemm(stream, xn, ctr_h, nullptr, nullptr, lscale, inv_sqrt_e,
              logits, nullptr, BT, Vv, Ee, 0, /*nt_store*/1);
}
